// TemporalSelfAttentionBlock_81312320848367
// MI455X (gfx1250) — matrix operand order repaired and checked by reading, not compiled, not run on hardware
//
// Fused temporal self-attention (B=2,T=16,C=32,H=W=192) for MI455X / gfx1250.
//
// Design (compile-only reasoning):
//  * AI ~32 FLOP/B vs 23.3 TB/s HBM -> matrix pipes required; bf16 WMMA with
//    f32 accumulate; x (151MB) is L2-resident (192MB L2).
//  * Fully fused single pass: x tile DMA'd in by TDM (tensor_load_to_lds),
//    result tile DMA'd out by TDM (tensor_store_from_lds). No scattered VMEM.
//  * T=16,C=32 -> every GEMM is exactly one v_wmma_f32_16x16x32_bf16 tile:
//    6 projection WMMAs + 1 score WMMA + 2 AV WMMAs = 9 per pixel.
//  * Cross-lane work on the VALU DPP path (row_xmask, permlanex16), not
//    ds_bpermute -> keeps the LDS pipe free for the operand relayouts.
//  * Q-projection bias dropped: softmax over j is invariant to the
//    j-constant term K_i . bq, so Q' = Wq x is exact. Saves VGPRs.

#include <hip/hip_runtime.h>
#include <hip/hip_bf16.h>

typedef __attribute__((ext_vector_type(16))) __bf16       v16bf;
typedef __attribute__((ext_vector_type(8)))  float        v8f;
typedef __attribute__((ext_vector_type(4)))  unsigned int u32x4;
typedef __attribute__((ext_vector_type(8)))  int          i32x8;
typedef __attribute__((ext_vector_type(4)))  int          i32x4;

static constexpr int TD  = 16;          // time steps
static constexpr int CD  = 32;          // channels
static constexpr int HW  = 192 * 192;   // pixels per (b,t,c) plane
static constexpr int PPW = 16;          // pixels per workgroup
static constexpr int NW  = 4;           // waves per workgroup
static constexpr int XROWSTRIDE = PPW + 1;  // 16 data dwords + 1 pad (TDM pad)

// Native f32->bf16 conversion (backend selects v_cvt_pk_bf16_f32 on gfx1250).
__device__ __forceinline__ __bf16 f2bf(float f) { return (__bf16)f; }

// lane ^ MASK within each 16-lane row, via DPP16 ROW_XMASK (one VALU op).
template <int MASK>
__device__ __forceinline__ float row_xor(float v) {
    int i = __builtin_bit_cast(int, v);
    int r = __builtin_amdgcn_update_dpp(i, i, 0x160 | MASK, 0xF, 0xF, true);
    return __builtin_bit_cast(float, r);
}

// lane ^ 16 (exchange between the two 16-lane halves) via v_permlanex16_b32.
__device__ __forceinline__ float xor16(float v) {
    int i = __builtin_bit_cast(int, v);
    int r = __builtin_amdgcn_permlanex16(i, i, 0x76543210, 0xFEDCBA98, false, false);
    return __builtin_bit_cast(float, r);
}

__global__ __launch_bounds__(128) void tsa_fused_kernel(
    const float* __restrict__ x,
    const float* __restrict__ Wk, const float* __restrict__ bk,
    const float* __restrict__ Wq, const float* __restrict__ bq,
    const float* __restrict__ Wv, const float* __restrict__ bv,
    float* __restrict__ out)
{
    // x tile: 512 rows (t*32+c) x 16 pixels, +1 dword pad per row (TDM pad_en)
    __shared__ float xtile[TD * CD * XROWSTRIDE];          // 34816 B
    __shared__ float scrK[NW][TD * 33];                    // K relayout / W reuse
    __shared__ float scrQ[NW][TD * 33];                    // Q relayout
    __shared__ float otile[TD * CD * PPW];                 // packed out tile, 32768 B

    const int tid   = threadIdx.x;
    const int lane  = tid & 31;
    const int wave  = tid >> 5;
    const int lhalf = lane >> 4;    // which 16-lane half
    const int l16   = lane & 15;

    const int tilesPerB = HW / PPW;
    const int b   = blockIdx.x / tilesPerB;
    const int p0  = (blockIdx.x % tilesPerB) * PPW;
    const size_t base = (size_t)b * TD * CD * HW + (size_t)p0;

    // ---------------- TDM: DMA the 512x16 f32 tile into LDS -----------------
    if (wave == 0) {
        unsigned long long ga = (unsigned long long)(const void*)(x + base);
        unsigned ldsOff = (unsigned)(size_t)(void*)&xtile[0];  // flat low 32 = LDS offset

        u32x4 g0;
        g0[0] = 1u;                                            // count=1, no gather
        g0[1] = ldsOff;                                        // lds_addr
        g0[2] = (unsigned)(ga & 0xFFFFFFFFull);                // global_addr[31:0]
        g0[3] = (unsigned)((ga >> 32) & 0x1FFFFFFull)          // global_addr[56:32]
              | (2u << 30);                                    // type=2 (image)

        i32x8 g1;
        // data_size=4B(2), pad_enable, pad_interval=16DW(3), pad_amount=1DW(0)
        g1[0] = (int)((2u << 16) | (1u << 20) | (3u << 22));
        g1[1] = (int)(((unsigned)HW & 0xFFFFu) << 16);         // tensor_dim0[15:0]
        g1[2] = (int)((((unsigned)HW >> 16) & 0xFFFFu)         // tensor_dim0[31:16]
              | ((unsigned)(TD * CD) << 16));                  // tensor_dim1[15:0]=512
        g1[3] = (int)((unsigned)PPW << 16);                    // tile_dim0=16 (dim1 hi=0)
        g1[4] = (int)(TD * CD);                                // tile_dim1=512, tile_dim2=0
        g1[5] = (int)HW;                                       // tensor_dim0_stride[31:0]
        g1[6] = 0;                                             // stride hi / dim1_stride lo
        g1[7] = 0;
        i32x4 z4 = {0, 0, 0, 0};                               // groups 2/3 unused (2D)
        i32x8 z8 = {0, 0, 0, 0, 0, 0, 0, 0};
        __builtin_amdgcn_tensor_load_to_lds(g0, g1, z4, z4, z8, 0);
        __builtin_amdgcn_s_wait_tensorcnt(0);
    }
    __syncthreads();

    // ------------- per-wave constant operands: Wk/Wq/Wv, biases -------------
    // B operand layout (32x16 bf16): lane L: N=c'=L%16(+tile), K rows per half:
    //   lanes 0-15 -> K=c=0..15 (elem e<->K=e), lanes 16-31 -> K=c=16..31.
    const float* Wm[3] = {Wk, Wq, Wv};
    v16bf Bw[3][2];
#pragma unroll
    for (int m = 0; m < 3; ++m) {
#pragma unroll
        for (int nb = 0; nb < 2; ++nb) {
            const int   cp  = nb * 16 + l16;                   // output channel c'
            const float* rw = Wm[m] + cp * CD + lhalf * 16;    // B[K=c,N=c']=W[c'][c]
            v16bf bb;
#pragma unroll
            for (int e = 0; e < 16; ++e) bb[e] = f2bf(rw[e]);
            Bw[m][nb] = bb;
        }
    }
    (void)bq;  // exact: K_i.bq is constant over softmax axis j -> drops out
    v8f biasK[2], biasV[2];
#pragma unroll
    for (int nb = 0; nb < 2; ++nb) {
        const float bsk = bk[nb * 16 + l16];
        const float bsv = bv[nb * 16 + l16];
        v8f k8, v8;
#pragma unroll
        for (int r = 0; r < 8; ++r) { k8[r] = bsk; v8[r] = bsv; }
        biasK[nb] = k8;
        biasV[nb] = v8;
    }

    float* myK = scrK[wave];
    float* myQ = scrQ[wave];
    const v8f vzero = {};

    // ------------------------- 4 pixels per wave ----------------------------
    for (int qq = 0; qq < PPW / NW; ++qq) {
        const int pl = wave * (PPW / NW) + qq;   // pixel within tile

        // A operand (16x32 bf16) layout: lane m<16 row t=m holds
        // K=0..15 (elem e<->K=e); lane m+16 same row holds K=16..31.
        v16bf aX;
        {
            const int t = l16;
#pragma unroll
            for (int e = 0; e < 16; ++e) {
                const int blk = (lhalf << 1) | (e >> 3);
                const int c   = blk * 8 + (e & 7);
                aX[e] = f2bf(xtile[(t * CD + c) * XROWSTRIDE + pl]);
            }
        }

        // --- projections: D[t,c'] = X(16x32) * W^T(32x16) (+bias via C) ---
        v8f accK[2], accQ[2], accV[2];
#pragma unroll
        for (int nb = 0; nb < 2; ++nb) {
            accK[nb] = __builtin_amdgcn_wmma_f32_16x16x32_bf16(false, aX, false, Bw[0][nb], (short)0, biasK[nb], false, false);
            accQ[nb] = __builtin_amdgcn_wmma_f32_16x16x32_bf16(false, aX, false, Bw[1][nb], (short)0, vzero,    false, false);
            accV[nb] = __builtin_amdgcn_wmma_f32_16x16x32_bf16(false, aX, false, Bw[2][nb], (short)0, biasV[nb], false, false);
        }

        // --- C-layout -> operand relayout for K (A) and Q (B) through LDS ---
#pragma unroll
        for (int nb = 0; nb < 2; ++nb) {
#pragma unroll
            for (int r = 0; r < 8; ++r) {
                const int t  = r + 8 * lhalf;
                const int cp = nb * 16 + l16;
                myK[t * 33 + cp] = accK[nb][r];
                myQ[t * 33 + cp] = accQ[nb][r];
            }
        }
        v16bf aK, bQ;
        {
            const int t = l16;
#pragma unroll
            for (int e = 0; e < 16; ++e) {
                const int blk = (lhalf << 1) | (e >> 3);
                const int c   = blk * 8 + (e & 7);
                aK[e] = f2bf(myK[t * 33 + c]);
            }
            const int j = l16;                 // B: N=j, K=c per half
#pragma unroll
            for (int e = 0; e < 16; ++e) {
                const int c = lhalf * 16 + e;
                bQ[e] = f2bf(myQ[j * 33 + c]);
            }
        }

        // --- scores: D[i,j] = K(16x32) * Q^T(32x16) ---
        v8f S = __builtin_amdgcn_wmma_f32_16x16x32_bf16(false, aK, false, bQ, (short)0, vzero, false, false);

        // --- softmax over j (= N = across each 16-lane half, DPP reductions) ---
        float w8[8];
#pragma unroll
        for (int r = 0; r < 8; ++r) {
            float v  = S[r];
            float mx = v;
            mx = fmaxf(mx, row_xor<1>(mx));
            mx = fmaxf(mx, row_xor<2>(mx));
            mx = fmaxf(mx, row_xor<4>(mx));
            mx = fmaxf(mx, row_xor<8>(mx));
            float e  = __expf(v - mx);
            float sm = e;
            sm += row_xor<1>(sm);
            sm += row_xor<2>(sm);
            sm += row_xor<4>(sm);
            sm += row_xor<8>(sm);
            w8[r] = e * __builtin_amdgcn_rcpf(sm);
        }

        // --- W (C-layout) -> A operand via LDS (reuse K scratch), pad K 16->32 ---
#pragma unroll
        for (int r = 0; r < 8; ++r) {
            const int i = r + 8 * lhalf;
            myK[i * 17 + l16] = w8[r];
        }
        v16bf aW;
        {
            const int i = l16;
#pragma unroll
            for (int e = 0; e < 16; ++e) {
                const int blk = (lhalf << 1) | (e >> 3);
                const int j   = blk * 8 + (e & 7);
                const float v = (j < 16) ? myK[i * 17 + j] : 0.0f;
                aW[e] = f2bf(v);
            }
        }

        // --- V (C-layout) -> B operand via permlanex16, K 16->32 pad ---
        v16bf bV[2];
#pragma unroll
        for (int nb = 0; nb < 2; ++nb) {
            v16bf bb;
#pragma unroll
            for (int r = 0; r < 8; ++r) {
                const float lo = accV[nb][r];    // j=r   @ this c'
                const float hi = xor16(lo);      // j=r+8 @ this c'
                bb[r]     = (lhalf == 0) ? f2bf(lo) : f2bf(0.0f);
                bb[r + 8] = (lhalf == 0) ? f2bf(hi) : f2bf(0.0f);
            }
            bV[nb] = bb;
        }

        // --- O = W * V, residual add, stage into packed LDS out tile ---
#pragma unroll
        for (int nb = 0; nb < 2; ++nb) {
            v8f O = __builtin_amdgcn_wmma_f32_16x16x32_bf16(false, aW, false, bV[nb], (short)0, vzero, false, false);
#pragma unroll
            for (int r = 0; r < 8; ++r) {
                const int t   = r + 8 * lhalf;
                const int c   = nb * 16 + l16;
                const int row = t * CD + c;
                otile[row * PPW + pl] = O[r] + xtile[row * XROWSTRIDE + pl];
            }
        }
    }

    // ---------------- TDM: DMA the packed 512x16 result tile out ------------
    __syncthreads();
    if (wave == 0) {
        unsigned long long ga = (unsigned long long)(void*)(out + base);
        unsigned ldsOff = (unsigned)(size_t)(void*)&otile[0];

        u32x4 g0;
        g0[0] = 1u;                                            // count=1
        g0[1] = ldsOff;                                        // lds_addr
        g0[2] = (unsigned)(ga & 0xFFFFFFFFull);
        g0[3] = (unsigned)((ga >> 32) & 0x1FFFFFFull) | (2u << 30);

        i32x8 g1;
        g1[0] = (int)(2u << 16);                               // data_size=4B, no pad
        g1[1] = (int)(((unsigned)HW & 0xFFFFu) << 16);         // tensor_dim0[15:0]
        g1[2] = (int)((((unsigned)HW >> 16) & 0xFFFFu)
              | ((unsigned)(TD * CD) << 16));                  // dim0 hi | dim1 lo
        g1[3] = (int)((unsigned)PPW << 16);                    // tile_dim0=16
        g1[4] = (int)(TD * CD);                                // tile_dim1=512
        g1[5] = (int)HW;                                       // tensor_dim0_stride
        g1[6] = 0;
        g1[7] = 0;
        i32x4 z4 = {0, 0, 0, 0};
        i32x8 z8 = {0, 0, 0, 0, 0, 0, 0, 0};
        __builtin_amdgcn_tensor_store_from_lds(g0, g1, z4, z4, z8, 0);
        __builtin_amdgcn_s_wait_tensorcnt(0);   // LDS must stay valid until done
    }
}

extern "C" void kernel_launch(void* const* d_in, const int* in_sizes, int n_in,
                              void* d_out, int out_size, void* d_ws, size_t ws_size,
                              hipStream_t stream) {
    const float* x  = (const float*)d_in[0];
    const float* Wk = (const float*)d_in[1];
    const float* bk = (const float*)d_in[2];
    const float* Wq = (const float*)d_in[3];
    const float* bq = (const float*)d_in[4];
    const float* Wv = (const float*)d_in[5];
    const float* bv = (const float*)d_in[6];
    float* out = (float*)d_out;
    (void)in_sizes; (void)n_in; (void)out_size; (void)d_ws; (void)ws_size;

    const int B = 2;
    dim3 grid(B * (HW / PPW));   // 4608 workgroups x 16 pixels
    dim3 block(128);             // 4 wave32
    hipLaunchKernelGGL(tsa_fused_kernel, grid, block, 0, stream,
                       x, Wk, bk, Wq, bq, Wv, bv, out);
}
